// TCMCProbability_75720273429152
// MI455X (gfx1250) — matrix operand order repaired and checked by reading, not compiled, not run on hardware
//
#include <hip/hip_runtime.h>
#include <hip/hip_bf16.h>

// ---------------------------------------------------------------------------
// Felsenstein pruning, 8 leaves / 64 states / 8 components / 100000 sites.
// ~92 GFLOP vs ~208 MB mandatory HBM traffic -> memory bound on MI455X
// (23.3 TB/s).  f16 WMMA (v_wmma_f32_16x16x32_f16) for all 14 per-site
// matvecs; P matrices precomputed in B-fragment layout; per-component
// fragment block (112KB) DMA'd into LDS with GLOBAL_LOAD_ASYNC_TO_LDS_B128
// and double-buffered (246KB of the 320KB WGP LDS) so staging overlaps WMMA.
// ---------------------------------------------------------------------------

typedef __attribute__((ext_vector_type(16))) _Float16 v16h;
typedef __attribute__((ext_vector_type(8)))  _Float16 v8h;
typedef __attribute__((ext_vector_type(8)))  float    v8f;
typedef __attribute__((ext_vector_type(4)))  float    v4f;

#define S64      64
#define MCOMP    8
#define NLEAVES  8
#define NEDGES   14
#define NSITES   100000
#define NTILES   (NSITES / 16)          // 6250, exact
#define WAVES_PB 8                      // 256-thread block = 8 wave32

#define PBUF_HALVES (NEDGES * 4096)     // 57344 halves = 114688 B per buffer

// dynamic LDS layout (bytes): pi x2 | PB fragments x2 | per-wave msg buffers
#define LDS_PI_OFF   0                              // 2 * 64 floats = 512 B
#define LDS_PB_OFF   512
#define LDS_MSG_OFF  (512 + 2 * 114688)             // 229888
#define LDS_TOTAL    (LDS_MSG_OFF + WAVES_PB * 2048)  // 246272 B

// ---------------------------------------------------------------------------
// Kernel 1: build Q (M,64,64) and pi (M,64).  One block per component.
// ---------------------------------------------------------------------------
__global__ void build_q_kernel(const float* __restrict__ rates,
                               const float* __restrict__ pi_inv,
                               float* __restrict__ Qout,
                               float* __restrict__ piOut) {
  const int m = blockIdx.x;
  const int tid = threadIdx.x;
  __shared__ float yv[64];
  __shared__ float piS[64];
  __shared__ float Qs[4096];
  __shared__ float rowS[64];
  __shared__ float nsS, invEmut;

  if (tid < 63) yv[tid] = pi_inv[m * 63 + tid];
  __syncthreads();
  if (tid == 0) {
    float ns = 0.f;
    for (int s = 0; s < 63; ++s) ns += yv[s] * yv[s];
    nsS = ns;
  }
  __syncthreads();
  if (tid < 64) {
    const float ns = nsS;
    const float v = (tid < 63) ? (2.f * yv[tid]) / (ns + 1.f)
                               : (ns - 1.f) / (ns + 1.f);
    piS[tid] = v * v;   // point on unit sphere squared -> sums to 1
  }
  __syncthreads();
  for (int idx = tid; idx < 4096; idx += 256) {
    const int i = idx >> 6, j = idx & 63;
    float q = 0.f;
    if (i != j) {
      const int a = (i < j) ? i : j;
      const int b = (i < j) ? j : i;
      const int tri = a * 64 - (a * (a + 1)) / 2 + (b - a - 1);
      const float r = rates[m * 2016 + tri];
      q = r * r * piS[j];
    }
    Qs[idx] = q;
  }
  __syncthreads();
  if (tid < 64) {
    float s = 0.f;
    for (int j = 0; j < 64; ++j) s += Qs[tid * 64 + j];
    rowS[tid] = s;
    Qs[tid * 64 + tid] = -s;
  }
  __syncthreads();
  if (tid == 0) {
    float e = 0.f;
    for (int i = 0; i < 64; ++i) e += piS[i] * rowS[i];
    invEmut = 1.f / e;
  }
  __syncthreads();
  const float inv = invEmut;
  for (int idx = tid; idx < 4096; idx += 256)
    Qout[(size_t)m * 4096 + idx] = Qs[idx] * inv;
  if (tid < 64) piOut[m * 64 + tid] = piS[tid];
}

// ---------------------------------------------------------------------------
// Kernel 2: P = expm(t*Q) per (edge, component); emit f16 B-fragments.
// Scaling-squaring (s=5) + order-8 Taylor Horner; ||tQ/32||inf ~ 0.07.
// B fragment (16x16x32 f16): lane L holds col N=L&15, K = kbase + (h<8 ? h : h+8),
// kbase = 32*kt + ((L&16)?8:0).  B[d,c] = P[c][d].
// ---------------------------------------------------------------------------
__global__ void expm_pb_kernel(const float* __restrict__ Q,
                               _Float16* __restrict__ gPB) {
  const int b = blockIdx.x;
  const int e = b >> 3;      // edge 0..13 (edge id == child node id)
  const int m = b & 7;
  const int tid = threadIdx.x;
  __shared__ float Bm[4096];
  __shared__ float R[4096];
  __shared__ float T[4096];

  const float t  = 0.05f * (float)(1 + (e & 3));   // TREE branch length
  const float sc = t * (1.0f / 32.0f);             // 2^-5 scaling
  for (int idx = tid; idx < 4096; idx += 256) {
    Bm[idx] = Q[(size_t)m * 4096 + idx] * sc;
    R[idx]  = ((idx >> 6) == (idx & 63)) ? 1.0f : 0.0f;
  }
  __syncthreads();
  // Horner Taylor: R = I + B*R/k, k = 8..1
  for (int k = 8; k >= 1; --k) {
    for (int idx = tid; idx < 4096; idx += 256) {
      const int i = idx >> 6, j = idx & 63;
      float s = 0.f;
#pragma unroll 8
      for (int kk = 0; kk < 64; ++kk) s += Bm[i * 64 + kk] * R[kk * 64 + j];
      T[idx] = s;
    }
    __syncthreads();
    const float ik = 1.0f / (float)k;
    for (int idx = tid; idx < 4096; idx += 256)
      R[idx] = T[idx] * ik + (((idx >> 6) == (idx & 63)) ? 1.0f : 0.0f);
    __syncthreads();
  }
  // 5 squarings
  for (int sq = 0; sq < 5; ++sq) {
    for (int idx = tid; idx < 4096; idx += 256) {
      const int i = idx >> 6, j = idx & 63;
      float s = 0.f;
#pragma unroll 8
      for (int kk = 0; kk < 64; ++kk) s += R[i * 64 + kk] * R[kk * 64 + j];
      T[idx] = s;
    }
    __syncthreads();
    for (int idx = tid; idx < 4096; idx += 256) R[idx] = T[idx];
    __syncthreads();
  }
  // emit B fragments: u = nt*1024 + kt*512 + lane*16 + h
  for (int u = tid; u < 4096; u += 256) {
    const int lane = (u >> 4) & 31, h = u & 15;
    const int nt = u >> 10, kt = (u >> 9) & 1;
    const int col = (nt << 4) + (lane & 15);
    const int d   = (kt << 5) + ((lane & 16) ? 8 : 0) + (h < 8 ? h : h + 8);
    gPB[(size_t)(e * 8 + m) * 4096 + u] = (_Float16)R[col * 64 + d];
  }
}

// ---------------------------------------------------------------------------
// Kernel 3: the Felsenstein site sweep with WMMA + async LDS staging.
// ---------------------------------------------------------------------------
struct AFrag { v16h lo; v16h hi; };   // K = 0..31 / 32..63 of state dim

// A fragment (16x32 f16): lane L = row (site) L&15; kbase=(L&16)?8:0;
// half h -> K = kbase + (h<8 ? h : h+8)
__device__ __forceinline__ v16h cvt_a_from_f32(const float* p, int kb) {
  const v4f a = *(const v4f*)(p + kb);
  const v4f b = *(const v4f*)(p + kb + 4);
  const v4f c = *(const v4f*)(p + kb + 16);
  const v4f d = *(const v4f*)(p + kb + 20);
  v16h r;
#pragma unroll
  for (int i = 0; i < 4; ++i) {
    r[i]      = (_Float16)a[i];
    r[4 + i]  = (_Float16)b[i];
    r[8 + i]  = (_Float16)c[i];
    r[12 + i] = (_Float16)d[i];
  }
  return r;
}

__device__ __forceinline__ AFrag load_leaf(const float* __restrict__ X,
                                           int i0, int leaf, int lane) {
  const int row = lane & 15;
  const int kb  = (lane & 16) ? 8 : 0;
  const float* p = X + (((size_t)(i0 + row) * NLEAVES + leaf) << 6);
  AFrag f;
  f.lo = cvt_a_from_f32(p, kb);
  f.hi = cvt_a_from_f32(p, kb + 32);
  return f;
}

__device__ __forceinline__ v8f edge_nt(const AFrag& a,
                                       const _Float16* __restrict__ pbuf,
                                       int e, int nt, int lane) {
  const _Float16* base = pbuf + (size_t)(((e << 2) + nt) << 1) * 512 + lane * 16;
  const v16h b0 = *(const v16h*)(base);          // kt = 0
  const v16h b1 = *(const v16h*)(base + 512);    // kt = 1
  v8f c = {};
  c = __builtin_amdgcn_wmma_f32_16x16x32_f16(false, a.lo, false, b0,
                                             (short)0, c, false, false);
  c = __builtin_amdgcn_wmma_f32_16x16x32_f16(false, a.hi, false, b1,
                                             (short)0, c, false, false);
  return c;
}

__device__ __forceinline__ void combine(const AFrag& a0, int e0,
                                        const AFrag& a1, int e1,
                                        const _Float16* __restrict__ pbuf,
                                        int lane, v8f prod[4]) {
#pragma unroll
  for (int nt = 0; nt < 4; ++nt) {
    const v8f c0 = edge_nt(a0, pbuf, e0, nt, lane);
    const v8f c1 = edge_nt(a1, pbuf, e1, nt, lane);
    prod[nt] = c0 * c1;   // Felsenstein product
  }
}

// D-layout (site-rows in VGPRs, state-cols in lanes) -> A-layout via LDS.
__device__ __forceinline__ AFrag roundtrip(const v8f prod[4],
                                           _Float16* msg, int lane) {
  const int rbase = (lane & 16) ? 8 : 0;
  const int cb    = lane & 15;
#pragma unroll
  for (int nt = 0; nt < 4; ++nt)
#pragma unroll
    for (int v = 0; v < 8; ++v)
      msg[(v + rbase) * 64 + nt * 16 + cb] = (_Float16)prod[nt][v];
  __asm__ volatile("s_wait_dscnt 0" ::: "memory");
  const int row = lane & 15;
  const int kb  = (lane & 16) ? 8 : 0;
  const _Float16* r = msg + row * 64;
  const v8h l0 = *(const v8h*)(r + kb);
  const v8h l1 = *(const v8h*)(r + kb + 16);
  const v8h h0 = *(const v8h*)(r + kb + 32);
  const v8h h1 = *(const v8h*)(r + kb + 48);
  AFrag f;
  f.lo = __builtin_shufflevector(l0, l1, 0,1,2,3,4,5,6,7,8,9,10,11,12,13,14,15);
  f.hi = __builtin_shufflevector(h0, h1, 0,1,2,3,4,5,6,7,8,9,10,11,12,13,14,15);
  return f;
}

// Stage one component's 112KB of B fragments global->LDS with the CDNA5
// async-to-LDS path (ASYNCcnt tracked, no VGPR round trip).  LDS address is
// the low 32 bits of the generic pointer.
__device__ __forceinline__ void stage_pb_async(const _Float16* __restrict__ gPB,
                                               _Float16* pbufDst, int m,
                                               int tid) {
#pragma unroll 4
  for (int c = tid; c < NEDGES * 512; c += 256) {   // 16B chunks
    const int e = c >> 9, w = c & 511;
    const unsigned long long g =
        (unsigned long long)(const void*)(gPB + (size_t)(e * 8 + m) * 4096 + w * 8);
    const unsigned int l =
        (unsigned int)(size_t)(const void*)(pbufDst + e * 4096 + w * 8);
    asm volatile("global_load_async_to_lds_b128 %0, %1, off"
                 :: "v"(l), "v"(g)
                 : "memory");
  }
}

__global__ void felsenstein_kernel(const float* __restrict__ X,
                                   const _Float16* __restrict__ gPB,
                                   const float* __restrict__ piG,
                                   float* __restrict__ out) {
  extern __shared__ char smem[];
  float*    piBuf = (float*)(smem + LDS_PI_OFF);       // 2 x 64 floats
  _Float16* pbuf  = (_Float16*)(smem + LDS_PB_OFF);    // 2 x PBUF_HALVES
  const int tid  = threadIdx.x;
  const int lane = tid & 31;
  const int wave = tid >> 5;
  _Float16* msg  = (_Float16*)(smem + LDS_MSG_OFF) + wave * 1024;

  const int tile = blockIdx.x * WAVES_PB + wave;

  // prologue: kick off component 0 staging
  stage_pb_async(gPB, pbuf, 0, tid);
  if (tid < 64) piBuf[tid] = piG[tid];

  for (int m = 0; m < MCOMP; ++m) {
    // wait for this component's async DMA (own wave), then make it visible
    asm volatile("s_wait_asynccnt 0" ::: "memory");
    __syncthreads();

    // overlap: start staging the NEXT component into the other buffer
    if (m + 1 < MCOMP) {
      stage_pb_async(gPB, pbuf + ((m + 1) & 1) * PBUF_HALVES, m + 1, tid);
      if (tid < 64) piBuf[((m + 1) & 1) * 64 + tid] = piG[(m + 1) * 64 + tid];
    }

    if (tile < NTILES) {
      const _Float16* pb  = pbuf + (m & 1) * PBUF_HALVES;
      const float*    piF = piBuf + (m & 1) * 64;
      const int i0 = tile << 4;
      v8f prod[4];

      AFrag a0 = load_leaf(X, i0, 0, lane);
      AFrag a1 = load_leaf(X, i0, 1, lane);
      combine(a0, 0, a1, 1, pb, lane, prod);
      const AFrag A8f = roundtrip(prod, msg, lane);

      a0 = load_leaf(X, i0, 2, lane);
      a1 = load_leaf(X, i0, 3, lane);
      combine(a0, 2, a1, 3, pb, lane, prod);
      const AFrag A9f = roundtrip(prod, msg, lane);

      combine(A8f, 8, A9f, 9, pb, lane, prod);
      const AFrag A12f = roundtrip(prod, msg, lane);

      a0 = load_leaf(X, i0, 4, lane);
      a1 = load_leaf(X, i0, 5, lane);
      combine(a0, 4, a1, 5, pb, lane, prod);
      const AFrag A10f = roundtrip(prod, msg, lane);

      a0 = load_leaf(X, i0, 6, lane);
      a1 = load_leaf(X, i0, 7, lane);
      combine(a0, 6, a1, 7, pb, lane, prod);
      const AFrag A11f = roundtrip(prod, msg, lane);

      combine(A10f, 10, A11f, 11, pb, lane, prod);
      const AFrag A13f = roundtrip(prod, msg, lane);

      combine(A12f, 12, A13f, 13, pb, lane, prod);   // root message

      // marginalize against pi: out[i,m] = sum_c A14[i,c] * pi[m,c]
      const int cb = lane & 15;
#pragma unroll
      for (int v = 0; v < 8; ++v) {
        float s = 0.f;
#pragma unroll
        for (int nt = 0; nt < 4; ++nt)
          s += prod[nt][v] * piF[nt * 16 + cb];
#pragma unroll
        for (int off = 1; off < 16; off <<= 1)
          s += __shfl_xor(s, off, 32);
        const int r = v + ((lane & 16) ? 8 : 0);
        if (cb == v)
          out[(size_t)(i0 + r) * MCOMP + m] = s;
      }
    }
  }
}

// ---------------------------------------------------------------------------
extern "C" void kernel_launch(void* const* d_in, const int* in_sizes, int n_in,
                              void* d_out, int out_size, void* d_ws,
                              size_t ws_size, hipStream_t stream) {
  const float* X      = (const float*)d_in[0];   // (100000, 8, 64)
  const float* rates  = (const float*)d_in[1];   // (8, 2016)
  const float* pi_inv = (const float*)d_in[2];   // (8, 63)
  float* out = (float*)d_out;                    // (100000, 8)

  // workspace: Q (128KB) | pi (2KB, +alignment) | PB fragments (896KB)
  float*    Q   = (float*)d_ws;
  float*    piG = (float*)((char*)d_ws + 131072);
  _Float16* gPB = (_Float16*)((char*)d_ws + 133120);

  build_q_kernel<<<MCOMP, 256, 0, stream>>>(rates, pi_inv, Q, piG);
  expm_pb_kernel<<<NEDGES * MCOMP, 256, 0, stream>>>(Q, gPB);

  hipFuncSetAttribute((const void*)felsenstein_kernel,
                      hipFuncAttributeMaxDynamicSharedMemorySize,
                      (int)LDS_TOTAL);
  const int blocks = (NTILES + WAVES_PB - 1) / WAVES_PB;   // 782
  felsenstein_kernel<<<blocks, 256, LDS_TOTAL, stream>>>(X, gPB, piG, out);
}
